// GridSemanticAggregator_79482664780148
// MI455X (gfx1250) — matrix operand order repaired and checked by reading, not compiled, not run on hardware
//
#include <hip/hip_runtime.h>
#include <hip/hip_bf16.h>

// ---------------------------------------------------------------------------
// Types for CDNA5 WMMA (gfx1250, wave32)
// ---------------------------------------------------------------------------
typedef __attribute__((ext_vector_type(8)))  unsigned short v8u;
typedef __attribute__((ext_vector_type(16))) unsigned short v16u;
typedef __attribute__((ext_vector_type(16))) __bf16        v16bf;
typedef __attribute__((ext_vector_type(8)))  float         v8f;

#define Bc   16
#define Mc   196
#define Nc   50
#define DGc  128
#define DSc  256
#define HLc  256
#define Hc   512
#define OUTc 512

// ---------------------------------------------------------------------------
// bf16 helpers (RNE convert via bit ops; storage type = unsigned short)
// ---------------------------------------------------------------------------
__device__ __forceinline__ unsigned short f2bf(float f) {
  union { float f; unsigned int u; } x; x.f = f;
  unsigned int u = x.u + 0x7fffu + ((x.u >> 16) & 1u);
  return (unsigned short)(u >> 16);
}
__device__ __forceinline__ float bf2f(unsigned short h) {
  union { unsigned int u; float f; } x; x.u = ((unsigned int)h) << 16;
  return x.f;
}

// A-matrix fragment (16x32 bf16), contiguous layout matching B:
// lane<16 : row=lane,    halfs 0..15 = K(k0+0..15)
// lane>=16: row=lane-16, halfs 0..15 = K(k0+16..31)
__device__ __forceinline__ v16bf load_a_frag(const unsigned short* base, int ld,
                                             int k0, int lane) {
  int r = lane & 15, hi = lane >> 4;
  const unsigned short* p = base + (size_t)r * ld + k0 + hi * 16;
  v8u lo = *(const v8u*)p;
  v8u hs = *(const v8u*)(p + 8);
  v16u v;
#pragma unroll
  for (int i = 0; i < 8; ++i) { v[i] = lo[i]; v[i + 8] = hs[i]; }
  return __builtin_bit_cast(v16bf, v);
}

// B-matrix fragment (32x16 bf16): lane = column; lanes 0-15 hold K k0+0..15,
// lanes 16-31 hold K k0+16..31 (16 contiguous halfs -> 32B load).
// W is row-major [C,K] (column n of B = row n of W).
__device__ __forceinline__ v16bf load_b_frag(const unsigned short* base, int ld,
                                             int k0, int lane) {
  int r = lane & 15, hi = lane >> 4;
  const unsigned short* p = base + (size_t)r * ld + k0 + hi * 16;
  v16u v = *(const v16u*)p;
  return __builtin_bit_cast(v16bf, v);
}

__device__ __forceinline__ v8f wmma_bf16(v16bf a, v16bf b, v8f c) {
  return __builtin_amdgcn_wmma_f32_16x16x32_bf16(false, a, false, b,
                                                 (short)0, c, false, false);
}

// ---------------------------------------------------------------------------
// fp32 -> bf16 conversion kernel
// ---------------------------------------------------------------------------
__global__ void cvt_f32_bf16(const float* __restrict__ s,
                             unsigned short* __restrict__ d, int n) {
  int i = blockIdx.x * blockDim.x + threadIdx.x;
  if (i < n) d[i] = f2bf(s[i]);
}

// ---------------------------------------------------------------------------
// GEMM:  Y[R,C] = act( X[R,K](bf16,lda) @ W[C,K](bf16,ldw)^T + bias )
// 4 column tiles (16x64 of output) per wave: the A fragment is loaded once
// per 32-K chunk and shared by 4 WMMAs. Requires R%16==0, K%32==0, C%64==0.
// Optional fp32 out, bf16 out, ReLU, accumulate-from-existing-Yf.
// ---------------------------------------------------------------------------
__global__ __launch_bounds__(256) void gemm_bias_act(
    const unsigned short* __restrict__ X, int lda,
    const unsigned short* __restrict__ W, int ldw,
    const float* __restrict__ bias,
    float* __restrict__ Yf, unsigned short* __restrict__ Yb,
    int R, int C, int K, int relu, int accum) {
  int tilesR = R >> 4, tilesC4 = C >> 6;
  int wid = (blockIdx.x << 3) + (threadIdx.x >> 5);
  if (wid >= tilesR * tilesC4) return;
  int tr = wid % tilesR, tc4 = wid / tilesR;
  int row0 = tr << 4, col0 = tc4 << 6;       // 64 output columns per wave
  int lane = threadIdx.x & 31;
  int nn = lane & 15, hi = lane >> 4;

  v8f acc[4];
  if (accum) {
#pragma unroll
    for (int t = 0; t < 4; ++t) {
      const float* yp = Yf + (size_t)(row0 + hi * 8) * C + col0 + t * 16 + nn;
#pragma unroll
      for (int v = 0; v < 8; ++v) acc[t][v] = yp[(size_t)v * C];
    }
  } else {
#pragma unroll
    for (int t = 0; t < 4; ++t)
#pragma unroll
      for (int v = 0; v < 8; ++v) acc[t][v] = 0.0f;
  }

  const unsigned short* xb = X + (size_t)row0 * lda;
  const unsigned short* wb = W + (size_t)col0 * ldw;
  for (int k0 = 0; k0 < K; k0 += 32) {
    v16bf a = load_a_frag(xb, lda, k0, lane);
#pragma unroll
    for (int t = 0; t < 4; ++t) {
      v16bf b = load_b_frag(wb + (size_t)(t * 16) * ldw, ldw, k0, lane);
      acc[t] = wmma_bf16(a, b, acc[t]);
    }
  }

#pragma unroll
  for (int t = 0; t < 4; ++t) {
    int col = col0 + t * 16 + nn;
    float bv = bias ? bias[col] : 0.0f;
    size_t obase = (size_t)(row0 + hi * 8) * C + col;
#pragma unroll
    for (int v = 0; v < 8; ++v) {
      float u = acc[t][v] + bv;
      if (relu && u < 0.0f) u = 0.0f;
      if (Yf) Yf[obase + (size_t)v * C] = u;
      if (Yb) Yb[obase + (size_t)v * C] = f2bf(u);
    }
  }
}

// ---------------------------------------------------------------------------
// BiLSTM: one block per direction, 1024 threads (32 waves).
// Per step: gates[16,1024] = x_t @ Wi^T + h @ Wh^T via WMMA (h in LDS bf16),
// 2 column tiles per wave sharing the A fragment; then elementwise LSTM cell
// (fp32 c in LDS), h written to LDS + global.
// ---------------------------------------------------------------------------
__global__ __launch_bounds__(1024) void lstm_kernel(
    const unsigned short* __restrict__ grid_bf,
    const unsigned short* __restrict__ wi_f, const unsigned short* __restrict__ wh_f,
    const float* __restrict__ bi_f, const float* __restrict__ bh_f,
    const unsigned short* __restrict__ wi_r, const unsigned short* __restrict__ wh_r,
    const float* __restrict__ bi_r, const float* __restrict__ bh_r,
    float* __restrict__ lstm_f, unsigned short* __restrict__ lstm_b) {
  __shared__ unsigned short hbf[16 * 256];   // 8 KB  (h, bf16)
  __shared__ float          cs [16 * 256];   // 16 KB (cell state, fp32)
  __shared__ unsigned short gbf[16 * 1024];  // 32 KB (gates, bf16)

  int dir = blockIdx.x;
  const unsigned short* wi = dir ? wi_r : wi_f;
  const unsigned short* wh = dir ? wh_r : wh_f;
  const float* bi = dir ? bi_r : bi_f;
  const float* bh = dir ? bh_r : bh_f;

  int tid = threadIdx.x;
  for (int i = tid; i < 4096; i += 1024) { hbf[i] = 0; cs[i] = 0.0f; }
  __syncthreads();

  int wid = tid >> 5, lane = tid & 31;
  int nn = lane & 15, hi = lane >> 4;
  int col0 = (wid * 2) << 4;                 // two adjacent 16-col tiles

  for (int step = 0; step < Mc; ++step) {
    int t = dir ? (Mc - 1 - step) : step;

    // ---- gate GEMM: 64 tiles of 16x16, 2 tiles per wave, shared A frag ----
    v8f acc0, acc1;
#pragma unroll
    for (int v = 0; v < 8; ++v) { acc0[v] = 0.0f; acc1[v] = 0.0f; }

    const unsigned short* xb  = grid_bf + (size_t)t * DGc;  // row stride M*DG
    const unsigned short* wi0 = wi + (size_t)col0 * DGc;
#pragma unroll
    for (int k0 = 0; k0 < DGc; k0 += 32) {
      v16bf a  = load_a_frag(xb, Mc * DGc, k0, lane);
      v16bf b0 = load_b_frag(wi0, DGc, k0, lane);
      v16bf b1 = load_b_frag(wi0 + (size_t)16 * DGc, DGc, k0, lane);
      acc0 = wmma_bf16(a, b0, acc0);
      acc1 = wmma_bf16(a, b1, acc1);
    }
    const unsigned short* wh0 = wh + (size_t)col0 * HLc;
#pragma unroll
    for (int k0 = 0; k0 < HLc; k0 += 32) {
      v16bf a  = load_a_frag(hbf, HLc, k0, lane);   // h from LDS
      v16bf b0 = load_b_frag(wh0, HLc, k0, lane);
      v16bf b1 = load_b_frag(wh0 + (size_t)16 * HLc, HLc, k0, lane);
      acc0 = wmma_bf16(a, b0, acc0);
      acc1 = wmma_bf16(a, b1, acc1);
    }
#pragma unroll
    for (int v = 0; v < 8; ++v) {
      gbf[(hi * 8 + v) * 1024 + col0 + nn]      = f2bf(acc0[v]);
      gbf[(hi * 8 + v) * 1024 + col0 + 16 + nn] = f2bf(acc1[v]);
    }
    __syncthreads();

    // ---- elementwise LSTM cell: 16 batches x 256 hidden ----
    for (int idx = tid; idx < 4096; idx += 1024) {
      int b = idx >> 8, j = idx & 255;
      const unsigned short* gr = gbf + b * 1024;
      float ig = bf2f(gr[j])       + bi[j]       + bh[j];
      float fg = bf2f(gr[256 + j]) + bi[256 + j] + bh[256 + j];
      float gg = bf2f(gr[512 + j]) + bi[512 + j] + bh[512 + j];
      float og = bf2f(gr[768 + j]) + bi[768 + j] + bh[768 + j];
      float si = 1.0f / (1.0f + __expf(-ig));
      float sf = 1.0f / (1.0f + __expf(-fg));
      float so = 1.0f / (1.0f + __expf(-og));
      float cv = sf * cs[idx] + si * tanhf(gg);
      float hv = so * tanhf(cv);
      cs[idx] = cv;
      unsigned short hb = f2bf(hv);
      hbf[idx] = hb;
      size_t o = ((size_t)b * Mc + t) * (2 * HLc) + (dir ? HLc : 0) + j;
      lstm_f[o] = hv;
      lstm_b[o] = hb;
    }
    __syncthreads();
  }
}

// ---------------------------------------------------------------------------
// Pair scores + softmax + attention context, fused. One block per (b, p).
//   score[q] = sum_h relu(A[b,p,h] + Bm[b,q,h] + cb1[h]) * cw2[h] + cb2
//   attn = softmax_q(score);  ctx[d] = sum_q attn[q] * V[b,q,d]  (d < 256)
// ---------------------------------------------------------------------------
__global__ __launch_bounds__(256) void pair_attn_ctx(
    const float* __restrict__ A, const float* __restrict__ Bm,
    const float* __restrict__ cb1, const float* __restrict__ cw2,
    const float* __restrict__ cb2p,
    const float* __restrict__ V, int ldv,
    unsigned short* __restrict__ ctx_bf,
    int P, int Q) {
  __shared__ float sa[Hc];
  __shared__ float sq[224];
  __shared__ float red[256];

  int b = blockIdx.x / P, p = blockIdx.x % P;
  int tid = threadIdx.x;
  const float* ar = A + ((size_t)b * P + p) * Hc;
  for (int h = tid; h < Hc; h += 256) sa[h] = ar[h] + cb1[h];
  __syncthreads();

  for (int q = 0; q < Q; ++q) {
    const float* br = Bm + ((size_t)b * Q + q) * Hc;
    float part = 0.0f;
    for (int h = tid; h < Hc; h += 256) {
      float tv = sa[h] + br[h];
      if (tv > 0.0f) part += tv * cw2[h];
    }
    red[tid] = part; __syncthreads();
    for (int s = 128; s > 0; s >>= 1) {
      if (tid < s) red[tid] += red[tid + s];
      __syncthreads();
    }
    if (tid == 0) sq[q] = red[0] + cb2p[0];
    __syncthreads();
  }

  // softmax over q
  float mx = -3.4e38f;
  for (int q = tid; q < Q; q += 256) mx = fmaxf(mx, sq[q]);
  red[tid] = mx; __syncthreads();
  for (int s = 128; s > 0; s >>= 1) {
    if (tid < s) red[tid] = fmaxf(red[tid], red[tid + s]);
    __syncthreads();
  }
  mx = red[0]; __syncthreads();
  float sm = 0.0f;
  for (int q = tid; q < Q; q += 256) {
    float e = __expf(sq[q] - mx);
    sq[q] = e; sm += e;
  }
  red[tid] = sm; __syncthreads();
  for (int s = 128; s > 0; s >>= 1) {
    if (tid < s) red[tid] += red[tid + s];
    __syncthreads();
  }
  float inv = 1.0f / red[0];
  __syncthreads();

  // context, d = tid (256 dims)
  int d = tid;
  float acc = 0.0f;
  for (int q = 0; q < Q; ++q)
    acc += sq[q] * V[((size_t)b * Q + q) * ldv + d];
  ctx_bf[((size_t)b * P + p) * DSc + d] = f2bf(acc * inv);
}

// ---------------------------------------------------------------------------
// Host-side launch
// ---------------------------------------------------------------------------
static void cvt(hipStream_t s, const float* src, unsigned short* dst, int n) {
  cvt_f32_bf16<<<(n + 255) / 256, 256, 0, s>>>(src, dst, n);
}
static void gemm(hipStream_t s, const unsigned short* X, int lda,
                 const unsigned short* W, int ldw, const float* bias,
                 float* Yf, unsigned short* Yb, int R, int C, int K,
                 int relu, int accum) {
  int tiles = (R / 16) * (C / 64);
  int blocks = (tiles + 7) / 8;
  gemm_bias_act<<<blocks, 256, 0, s>>>(X, lda, W, ldw, bias, Yf, Yb,
                                       R, C, K, relu, accum);
}

extern "C" void kernel_launch(void* const* d_in, const int* in_sizes, int n_in,
                              void* d_out, int out_size, void* d_ws, size_t ws_size,
                              hipStream_t stream) {
  // Input order: grid_features, semantic_features, semantic_adj, then params
  // dict in insertion order.
  const float* grid = (const float*)d_in[0];
  const float* sem  = (const float*)d_in[1];
#define P_(i) ((const float*)d_in[3 + (i)])
  const float* p_lstm_wif = P_(0),  *p_lstm_whf = P_(1);
  const float* p_lstm_bif = P_(2),  *p_lstm_bhf = P_(3);
  const float* p_lstm_wir = P_(4),  *p_lstm_whr = P_(5);
  const float* p_lstm_bir = P_(6),  *p_lstm_bhr = P_(7);
  const float* p_fs_w1 = P_(8),  *p_fs_b1 = P_(9),  *p_fs_w2 = P_(10), *p_fs_b2 = P_(11);
  const float* p_fc_w1 = P_(12), *p_fc_b1 = P_(13), *p_fc_w2 = P_(14), *p_fc_b2 = P_(15);
  const float* p_fv_w1 = P_(16), *p_fv_b1 = P_(17), *p_fv_w2 = P_(18), *p_fv_b2 = P_(19);
  const float* p_fh_w1 = P_(20), *p_fh_b1 = P_(21), *p_fh_w2 = P_(22), *p_fh_b2 = P_(23);
  const float* p_fg_w1 = P_(24), *p_fg_b1 = P_(25), *p_fg_w2 = P_(26), *p_fg_b2 = P_(27);
  const float* p_corr_w1 = P_(28), *p_corr_b1 = P_(29);
  const float* p_corr_w2 = P_(30), *p_corr_b2 = P_(31);
  const float* p_semp_w = P_(32), *p_semp_b = P_(33);
  const float* p_visp_w = P_(34), *p_visp_b = P_(35);
#undef P_

  const int RM = Bc * Mc;   // 3136
  const int RN = Bc * Nc;   // 800

  // Bump allocator over d_ws
  size_t cur = 0;
  auto alloc = [&](size_t bytes) -> char* {
    char* p = (char*)d_ws + cur;
    cur += (bytes + 255) & ~(size_t)255;
    return p;
  };
  auto au16 = [&](size_t n) -> unsigned short* { return (unsigned short*)alloc(n * 2); };
  auto af32 = [&](size_t n) -> float*          { return (float*)alloc(n * 4); };

  // bf16 weight copies
  unsigned short* wif_bf = au16(4 * HLc * DGc);
  unsigned short* whf_bf = au16(4 * HLc * HLc);
  unsigned short* wir_bf = au16(4 * HLc * DGc);
  unsigned short* whr_bf = au16(4 * HLc * HLc);
  unsigned short* fs1_bf = au16(Hc * DSc);
  unsigned short* fs2_bf = au16(Hc * Hc);
  unsigned short* fc1_bf = au16(Hc * 2 * HLc);
  unsigned short* fc2_bf = au16(Hc * Hc);
  unsigned short* fv1_bf = au16(Hc * DGc);
  unsigned short* fv2_bf = au16(Hc * Hc);
  unsigned short* fh1_bf = au16(Hc * DSc);
  unsigned short* fh2_bf = au16(Hc * Hc);
  unsigned short* fg1_bf = au16(Hc * DSc);
  unsigned short* fg2_bf = au16(Hc * Hc);
  unsigned short* cw1_bf = au16(Hc * 2 * Hc);
  unsigned short* semp_bf = au16(OUTc * (DSc + Hc));
  unsigned short* visp_bf = au16(OUTc * (DGc + Hc));

  // bf16 activations
  unsigned short* grid_bf = au16((size_t)RM * DGc);
  unsigned short* sem_bf  = au16((size_t)RN * DSc);
  unsigned short* lstm_bf = au16((size_t)RM * 2 * HLc);
  unsigned short* tmp_bf  = au16((size_t)RM * Hc);      // shared MLP hidden
  unsigned short* es_bf   = au16((size_t)RN * Hc);
  unsigned short* ec_bf   = au16((size_t)RM * Hc);
  unsigned short* ev_bf   = au16((size_t)RM * Hc);
  unsigned short* eg_bf   = au16((size_t)RN * Hc);
  unsigned short* evc_bf  = au16((size_t)RN * Hc);
  unsigned short* vctx_bf = au16((size_t)RN * DSc);
  unsigned short* sctx_bf = au16((size_t)RM * DSc);
  unsigned short* esc_bf  = ec_bf;                      // alias (ec dead by then)

  // fp32 activations
  float* lstm_f = af32((size_t)RM * 2 * HLc);
  float* A1  = af32((size_t)RN * Hc);
  float* Bm1 = af32((size_t)RM * Hc);
  float* A2  = Bm1;   // alias (Bm1 dead after pair1)
  float* Bm2 = A1;    // alias (A1 dead after pair1)

  float* out_refined  = (float*)d_out;                   // (B,N,OUT)
  float* out_enhanced = out_refined + (size_t)RN * OUTc; // (B,M,OUT)

  // ---- convert inputs & weights to bf16 ----
  cvt(stream, grid, grid_bf, RM * DGc);
  cvt(stream, sem,  sem_bf,  RN * DSc);
  cvt(stream, p_lstm_wif, wif_bf, 4 * HLc * DGc);
  cvt(stream, p_lstm_whf, whf_bf, 4 * HLc * HLc);
  cvt(stream, p_lstm_wir, wir_bf, 4 * HLc * DGc);
  cvt(stream, p_lstm_whr, whr_bf, 4 * HLc * HLc);
  cvt(stream, p_fs_w1, fs1_bf, Hc * DSc);
  cvt(stream, p_fs_w2, fs2_bf, Hc * Hc);
  cvt(stream, p_fc_w1, fc1_bf, Hc * 2 * HLc);
  cvt(stream, p_fc_w2, fc2_bf, Hc * Hc);
  cvt(stream, p_fv_w1, fv1_bf, Hc * DGc);
  cvt(stream, p_fv_w2, fv2_bf, Hc * Hc);
  cvt(stream, p_fh_w1, fh1_bf, Hc * DSc);
  cvt(stream, p_fh_w2, fh2_bf, Hc * Hc);
  cvt(stream, p_fg_w1, fg1_bf, Hc * DSc);
  cvt(stream, p_fg_w2, fg2_bf, Hc * Hc);
  cvt(stream, p_corr_w1, cw1_bf, Hc * 2 * Hc);
  cvt(stream, p_semp_w, semp_bf, OUTc * (DSc + Hc));
  cvt(stream, p_visp_w, visp_bf, OUTc * (DGc + Hc));

  // ---- BiLSTM (2 blocks: fwd / rev) ----
  lstm_kernel<<<2, 1024, 0, stream>>>(grid_bf,
      wif_bf, whf_bf, p_lstm_bif, p_lstm_bhf,
      wir_bf, whr_bf, p_lstm_bir, p_lstm_bhr,
      lstm_f, lstm_bf);

  // ---- es = mlp2(sem) ; ec = mlp2(lstm_feats) ----
  gemm(stream, sem_bf, DSc, fs1_bf, DSc, p_fs_b1, nullptr, tmp_bf, RN, Hc, DSc, 1, 0);
  gemm(stream, tmp_bf, Hc,  fs2_bf, Hc,  p_fs_b2, nullptr, es_bf,  RN, Hc, Hc,  1, 0);
  gemm(stream, lstm_bf, 2 * HLc, fc1_bf, 2 * HLc, p_fc_b1, nullptr, tmp_bf, RM, Hc, 2 * HLc, 1, 0);
  gemm(stream, tmp_bf, Hc, fc2_bf, Hc, p_fc_b2, nullptr, ec_bf, RM, Hc, Hc, 1, 0);

  // ---- A1 = es @ wa^T, Bm1 = ec @ wb^T  (corr_w1 split by columns) ----
  gemm(stream, es_bf, Hc, cw1_bf,      2 * Hc, nullptr, A1,  nullptr, RN, Hc, Hc, 0, 0);
  gemm(stream, ec_bf, Hc, cw1_bf + Hc, 2 * Hc, nullptr, Bm1, nullptr, RM, Hc, Hc, 0, 0);

  // ---- pair1: scores + softmax over M + visual_ctx[..., :DS] ----
  pair_attn_ctx<<<Bc * Nc, 256, 0, stream>>>(A1, Bm1, p_corr_b1, p_corr_w2,
      p_corr_b2, lstm_f, 2 * HLc, vctx_bf, Nc, Mc);

  // ---- evc = mlp2(visual_ctx[:,:,:DS]) ----
  gemm(stream, vctx_bf, DSc, fh1_bf, DSc, p_fh_b1, nullptr, tmp_bf, RN, Hc, DSc, 1, 0);
  gemm(stream, tmp_bf, Hc, fh2_bf, Hc, p_fh_b2, nullptr, evc_bf, RN, Hc, Hc, 1, 0);

  // ---- refined = [sem, evc] @ semp_w^T + semp_b  (two passes, accumulate) ----
  gemm(stream, sem_bf, DSc, semp_bf, DSc + Hc, p_semp_b, out_refined, nullptr,
       RN, OUTc, DSc, 0, 0);
  gemm(stream, evc_bf, Hc, semp_bf + DSc, DSc + Hc, nullptr, out_refined, nullptr,
       RN, OUTc, Hc, 0, 1);

  // ---- ev = mlp2(grid) ; eg = mlp2(sem) ----
  gemm(stream, grid_bf, DGc, fv1_bf, DGc, p_fv_b1, nullptr, tmp_bf, RM, Hc, DGc, 1, 0);
  gemm(stream, tmp_bf, Hc, fv2_bf, Hc, p_fv_b2, nullptr, ev_bf, RM, Hc, Hc, 1, 0);
  gemm(stream, sem_bf, DSc, fg1_bf, DSc, p_fg_b1, nullptr, tmp_bf, RN, Hc, DSc, 1, 0);
  gemm(stream, tmp_bf, Hc, fg2_bf, Hc, p_fg_b2, nullptr, eg_bf, RN, Hc, Hc, 1, 0);

  // ---- A2 = ev @ wa^T, Bm2 = eg @ wb^T ----
  gemm(stream, ev_bf, Hc, cw1_bf,      2 * Hc, nullptr, A2,  nullptr, RM, Hc, Hc, 0, 0);
  gemm(stream, eg_bf, Hc, cw1_bf + Hc, 2 * Hc, nullptr, Bm2, nullptr, RN, Hc, Hc, 0, 0);

  // ---- pair2: scores + softmax over N + semantic_ctx ----
  pair_attn_ctx<<<Bc * Mc, 256, 0, stream>>>(A2, Bm2, p_corr_b1, p_corr_w2,
      p_corr_b2, sem, DSc, sctx_bf, Mc, Nc);

  // ---- esc = mlp2(semantic_ctx) with fg weights ----
  gemm(stream, sctx_bf, DSc, fg1_bf, DSc, p_fg_b1, nullptr, tmp_bf, RM, Hc, DSc, 1, 0);
  gemm(stream, tmp_bf, Hc, fg2_bf, Hc, p_fg_b2, nullptr, esc_bf, RM, Hc, Hc, 1, 0);

  // ---- enhanced = [grid, esc] @ visp_w^T + visp_b ----
  gemm(stream, grid_bf, DGc, visp_bf, DGc + Hc, p_visp_b, out_enhanced, nullptr,
       RM, OUTc, DGc, 0, 0);
  gemm(stream, esc_bf, Hc, visp_bf + DGc, DGc + Hc, nullptr, out_enhanced, nullptr,
       RM, OUTc, Hc, 0, 1);

  (void)in_sizes; (void)n_in; (void)out_size; (void)ws_size;
}
